// InterestEvolving_11836929868183
// MI455X (gfx1250) — matrix operand order repaired and checked by reading, not compiled, not run on hardware
//
#include <hip/hip_runtime.h>
#include <hip/hip_bf16.h>
#include <cstddef>

typedef __attribute__((ext_vector_type(16))) _Float16 v16h;
typedef __attribute__((ext_vector_type(8)))  _Float16 f16x8;
typedef __attribute__((ext_vector_type(8)))  float    v8f;

#define LSEQ 200
#define DDIM 128
#define STRIDE 136   // halves per f16 LDS row: 128 + 8 pad -> 272B, 16B aligned
#define USTRIDE 132  // floats per f32 LDS row for u-buffer

// ---------------- Kernel A: attention scores + softmax ----------------
__global__ __launch_bounds__(256) void ie_attn_kernel(
    const float* __restrict__ h_states, const float* __restrict__ target,
    const unsigned char* __restrict__ mask, float* __restrict__ attn)
{
  const int b = blockIdx.x;
  const int tid = threadIdx.x;
  __shared__ float tgt[DDIM];
  __shared__ float sc[256];

  if (tid < DDIM) tgt[tid] = target[(size_t)b * DDIM + tid];
  __syncthreads();

  float s = -3.0e38f;
  if (tid < LSEQ) {
    const float* row = h_states + ((size_t)b * LSEQ + tid) * DDIM;
    float acc = 0.f;
    #pragma unroll 8
    for (int d = 0; d < DDIM; ++d) acc = fmaf(row[d], tgt[d], acc);
    acc *= 0.08838834764831845f;                 // 1/sqrt(128)
    if (!mask[(size_t)b * LSEQ + tid]) acc = -1.0e9f;
    s = acc;
  }
  sc[tid] = s;
  __syncthreads();
  #pragma unroll
  for (int off = 128; off > 0; off >>= 1) {
    if (tid < off) sc[tid] = fmaxf(sc[tid], sc[tid + off]);
    __syncthreads();
  }
  const float mx = sc[0];
  __syncthreads();
  const float e = (tid < LSEQ) ? __expf(s - mx) : 0.f;
  sc[tid] = e;
  __syncthreads();
  #pragma unroll
  for (int off = 128; off > 0; off >>= 1) {
    if (tid < off) sc[tid] += sc[tid + off];
    __syncthreads();
  }
  const float inv = 1.f / sc[0];
  if (tid < LSEQ) attn[(size_t)b * LSEQ + tid] = e * inv;
}

// ---------------- fragment builders ----------------
// B-matrix 32x16 (KxN) f16 layout: lanes 0-15 hold K=0..7,16..23 (col = lane),
// lanes 16-31 hold K=8..15,24..31 (same k order as the A fragment).
__device__ __forceinline__ v16h load_bfrag(const float* __restrict__ W,
                                           int kbase, int n, int half)
{
  v16h bf;
  #pragma unroll
  for (int v = 0; v < 8; ++v) {
    const int k = kbase + 8 * half + 2 * v + (v >= 4 ? 8 : 0);
    bf[2 * v]     = (_Float16)W[(size_t)k * DDIM + n];
    bf[2 * v + 1] = (_Float16)W[(size_t)(k + 1) * DDIM + n];
  }
  return bf;
}

// A-matrix 16x32 (MxK) f16 layout: lanes 0-15 row M=lane hold K 0..7 (v0-3)
// and K 16..23 (v4-7); lanes 16-31 hold K 8..15 / 24..31. Row-major LDS ->
// two 16B ds_load_b128 per fragment.
__device__ __forceinline__ v16h load_afrag(const _Float16* __restrict__ buf,
                                           int kbase, int m, int half)
{
  const _Float16* p = buf + m * STRIDE + kbase + 8 * half;
  const f16x8 lo = *(const f16x8*)(p);
  const f16x8 hi = *(const f16x8*)(p + 16);
  v16h a;
  #pragma unroll
  for (int i = 0; i < 8; ++i) { a[i] = lo[i]; a[8 + i] = hi[i]; }
  return a;
}

__device__ __forceinline__ float sigmoid_f(float x) {
  return 1.f / (1.f + __expf(-x));
}
__device__ __forceinline__ float tanh_f(float x) {
  const float t = __expf(2.f * x);
  return (t - 1.f) / (t + 1.f);
}

#define WMMA_F16(A, Bf, C) \
  __builtin_amdgcn_wmma_f32_16x16x32_f16(false, (A), false, (Bf), (short)0, (C), false, false)

// ---------------- Kernel B: sequential gated recurrence ----------------
// 768 threads = 24 waves: waves 0-7 -> r gate, 8-15 -> u gate, 16-23 -> h~.
// Wave (g, nt) owns output columns 16*nt .. 16*nt+15 of its gate.
__global__ __launch_bounds__(768, 1) void ie_scan_kernel(
    const float* __restrict__ h_states, const unsigned char* __restrict__ mask,
    const float* __restrict__ attn,
    const float* __restrict__ W_r, const float* __restrict__ b_r,
    const float* __restrict__ W_u, const float* __restrict__ b_u,
    const float* __restrict__ W_h, const float* __restrict__ b_h,
    float* __restrict__ out)
{
  __shared__ __align__(16) _Float16 sx [16 * STRIDE];   // x_t   (f16)
  __shared__ __align__(16) _Float16 sh [16 * STRIDE];   // h     (f16)
  __shared__ __align__(16) _Float16 srh[16 * STRIDE];   // r*h   (f16)
  __shared__ float su [16 * USTRIDE];                   // u     (f32)
  __shared__ float se [2][16];                          // attn e_t (dbl-buf)
  __shared__ float smk[2][16];                          // mask 0/1 (dbl-buf)

  const int tid  = threadIdx.x;
  const int ln   = tid & 31;
  const int m    = ln & 15;
  const int half = ln >> 4;
  const int b0   = blockIdx.x * 16;

  // wave-uniform role values: force into SGPRs so gate branches are scalar
  // (s_cbranch), keeping EXEC all-ones around every WMMA as the ISA requires.
  const int wu = __builtin_amdgcn_readfirstlane(tid >> 5);  // wave id 0..23
  const int gu = wu >> 3;                                   // 0 = r, 1 = u, 2 = h~
  const int nt = wu & 7;                                    // N-tile within gate
  const int n  = nt * 16 + m;                               // B-frag / C column

  // --- one-time: this wave's gate weights into registers (64 VGPRs) ---
  const float* Wg = (gu == 0) ? W_r : (gu == 1) ? W_u : W_h;
  const float* bg = (gu == 0) ? b_r : (gu == 1) ? b_u : b_h;
  v16h bw[8];
  #pragma unroll
  for (int t = 0; t < 8; ++t) bw[t] = load_bfrag(Wg, 32 * t, n, half);
  const float bias = bg[n];

  v8f hreg = {};                    // h~ waves: h slice, C layout row = v+8*half

  for (int i = tid; i < 16 * STRIDE; i += 768) sh[i] = (_Float16)0.f;

  // --- prologue: stage x_0, e_0, m_0 (waves 0-7 = 256 lanes) ---
  if (wu < 8) {
    const int r  = tid >> 4;
    const int cs = (tid & 15) * 8;
    const float* gx = h_states + ((size_t)(b0 + r) * LSEQ + 0) * DDIM + cs;
    f16x8 xv;
    #pragma unroll
    for (int i = 0; i < 8; ++i) xv[i] = (_Float16)gx[i];
    *(f16x8*)(sx + r * STRIDE + cs) = xv;
    if (tid < 16) {
      se [0][tid] = attn[(size_t)(b0 + tid) * LSEQ + 0];
      smk[0][tid] = mask[(size_t)(b0 + tid) * LSEQ + 0] ? 1.f : 0.f;
    }
  }
  __syncthreads();

  for (int l = 0; l < LSEQ; ++l) {
    const int buf = l & 1;

    // ---------------- phase A ----------------
    v16h ax[4];
    #pragma unroll
    for (int t = 0; t < 4; ++t) ax[t] = load_afrag(sx, 32 * t, m, half);

    v8f c = {};
    if (gu < 2) {
      // r / u gates: full K = 256 = [x | h]
      v16h ah[4];
      #pragma unroll
      for (int t = 0; t < 4; ++t) ah[t] = load_afrag(sh, 32 * t, m, half);
      #pragma unroll
      for (int t = 0; t < 4; ++t) c = WMMA_F16(ax[t], bw[t], c);
      #pragma unroll
      for (int t = 0; t < 4; ++t) c = WMMA_F16(ah[t], bw[4 + t], c);
      if (gu == 0) {
        // publish r*h (h read back from f16 h buffer)
        #pragma unroll
        for (int v = 0; v < 8; ++v) {
          const int row = v + 8 * half;
          const float hv = (float)sh[row * STRIDE + n];
          const float rv = sigmoid_f(c[v] + bias);
          srh[row * STRIDE + n] = (_Float16)(rv * hv);
        }
      } else {
        // publish u
        #pragma unroll
        for (int v = 0; v < 8; ++v) {
          const int row = v + 8 * half;
          su[row * USTRIDE + n] = sigmoid_f(c[v] + bias);
        }
      }
    } else {
      // h~ gate: x-part only (r*h part after the barrier)
      #pragma unroll
      for (int t = 0; t < 4; ++t) c = WMMA_F16(ax[t], bw[t], c);
    }
    __syncthreads();   // b2: srh / su published

    // ---------------- phase B ----------------
    if (gu == 2) {
      v16h arh[4];
      #pragma unroll
      for (int t = 0; t < 4; ++t) arh[t] = load_afrag(srh, 32 * t, m, half);
      #pragma unroll
      for (int t = 0; t < 4; ++t) c = WMMA_F16(arh[t], bw[4 + t], c);
      #pragma unroll
      for (int v = 0; v < 8; ++v) {
        const int row = v + 8 * half;
        const float ev = se [buf][row];
        const float mk = smk[buf][row];
        const float ht = tanh_f(c[v] + bias);
        const float uv = su[row * USTRIDE + n];
        const float up = ev * uv;
        const float hn = (1.f - up) * hreg[v] + up * ht;
        hreg[v] = mk * hn + (1.f - mk) * hreg[v];
        sh[row * STRIDE + n] = (_Float16)hreg[v];
      }
    } else if (gu == 0 && l + 1 < LSEQ) {
      // idle r-waves (256 lanes) software-pipeline the next step's inputs
      const int r  = tid >> 4;
      const int cs = (tid & 15) * 8;
      const float* gx = h_states + ((size_t)(b0 + r) * LSEQ + (l + 1)) * DDIM + cs;
      f16x8 xv;
      #pragma unroll
      for (int i = 0; i < 8; ++i) xv[i] = (_Float16)gx[i];
      *(f16x8*)(sx + r * STRIDE + cs) = xv;
      if (tid < 16) {
        se [buf ^ 1][tid] = attn[(size_t)(b0 + tid) * LSEQ + (l + 1)];
        smk[buf ^ 1][tid] = mask[(size_t)(b0 + tid) * LSEQ + (l + 1)] ? 1.f : 0.f;
      }
    }
    __syncthreads();   // b1 of next step: sh / sx / se ready
  }

  // --- final h written by h~ waves ---
  if (gu == 2) {
    #pragma unroll
    for (int v = 0; v < 8; ++v) {
      const int row = v + 8 * half;
      out[(size_t)(b0 + row) * DDIM + n] = hreg[v];
    }
  }
}

extern "C" void kernel_launch(void* const* d_in, const int* in_sizes, int n_in,
                              void* d_out, int out_size, void* d_ws, size_t ws_size,
                              hipStream_t stream) {
  const float*         h_states = (const float*)d_in[0];
  const float*         target   = (const float*)d_in[1];
  const unsigned char* maskp    = (const unsigned char*)d_in[2];
  const float*         W_r      = (const float*)d_in[3];
  const float*         b_r      = (const float*)d_in[4];
  const float*         W_u      = (const float*)d_in[5];
  const float*         b_u      = (const float*)d_in[6];
  const float*         W_h      = (const float*)d_in[7];
  const float*         b_h      = (const float*)d_in[8];
  float*               out      = (float*)d_out;

  const int Bn = in_sizes[1] / DDIM;      // 4096
  float* attn = (float*)d_ws;             // Bn * LSEQ floats

  ie_attn_kernel<<<Bn, 256, 0, stream>>>(h_states, target, maskp, attn);
  ie_scan_kernel<<<Bn / 16, 768, 0, stream>>>(h_states, maskp, attn,
                                              W_r, b_r, W_u, b_u, W_h, b_h, out);
}
